// CascadeModel_35270271435253
// MI455X (gfx1250) — matrix operand order repaired and checked by reading, not compiled, not run on hardware
//
#include <hip/hip_runtime.h>
#include <hip/hip_bf16.h>

typedef __attribute__((ext_vector_type(16))) __bf16 v16bf;
typedef __attribute__((ext_vector_type(8)))  float  v8f;
typedef __attribute__((ext_vector_type(2)))  float  v2f;

constexpr int kH    = 256;
constexpr int kV    = 32000;
constexpr int kB    = 32;
constexpr int kT    = 1024;
constexpr int kFAST = 64;
constexpr int kSLOW = 256;
#define NEGV (-1e9f)

union BfFrag { v16bf v; unsigned int u[8]; };

__device__ __forceinline__ unsigned int f2bf(float x) {
  unsigned int u = __float_as_uint(x);
  return (u + 0x7fffu + ((u >> 16) & 1u)) >> 16;   // RNE f32 -> bf16 bits
}

__device__ __forceinline__ float waveRedSum(float v) {
#pragma unroll
  for (int off = 16; off > 0; off >>= 1) v += __shfl_xor(v, off, 32);
  return v;
}

// ---------------------------------------------------------------------------
// K1: h0 = embed[seq]; also keep bf16 copy for WMMA A-matrix.
// ---------------------------------------------------------------------------
__global__ void __launch_bounds__(kH)
k_gather(const int* __restrict__ seq, const float* __restrict__ embed,
         float* __restrict__ h0f, unsigned short* __restrict__ h0b) {
  const int bt  = blockIdx.x;        // 0..B*T-1
  const int col = threadIdx.x;       // 0..255
  const int tok = seq[bt];
  const float v = embed[(size_t)tok * kH + col];
  const size_t o = (size_t)bt * kH + col;
  h0f[o] = v;
  h0b[o] = (unsigned short)f2bf(v);
}

// ---------------------------------------------------------------------------
// K2: pack W1 [256,512] and W2 [512,256] into WMMA bf16 B-fragment layout:
//   u32 index ((nt*KS + ks)*32 + lane)*8 + i  holds  bf16(W[k][n]) | bf16(W[k+1][n])<<16
//   with n = nt*16 + (lane&15),  k = ks*32 + (lane>>4)*16 + 2*i
// so each lane's fragment is 8 contiguous dwords (fully coalesced loads).
// ---------------------------------------------------------------------------
__global__ void __launch_bounds__(256)
k_pack(const float* __restrict__ W1, const float* __restrict__ W2,
       unsigned int* __restrict__ W1p, unsigned int* __restrict__ W2p) {
  const int tid = blockIdx.x * 256 + threadIdx.x;   // 0..131071
  if (tid < 65536) {                                // W1: NT=32, KS=8, N=512
    const int i = tid & 7, lane = (tid >> 3) & 31, ks = (tid >> 8) & 7, nt = tid >> 11;
    const int n = nt * 16 + (lane & 15);
    const int k = ks * 32 + (lane >> 4) * 16 + 2 * i;
    W1p[tid] = f2bf(W1[k * 512 + n]) | (f2bf(W1[(k + 1) * 512 + n]) << 16);
  } else {                                          // W2: NT=16, KS=16, N=256
    const int t = tid - 65536;
    const int i = t & 7, lane = (t >> 3) & 31, ks = (t >> 8) & 15, nt = t >> 12;
    const int n = nt * 16 + (lane & 15);
    const int k = ks * 32 + (lane >> 4) * 16 + 2 * i;
    W2p[t] = f2bf(W2[k * 256 + n]) | (f2bf(W2[(k + 1) * 256 + n]) << 16);
  }
}

// A-fragment (16x32 bf16) load from a row-major LDS tile [16][ldk] (bf16).
__device__ __forceinline__ void loadA(BfFrag& a, const unsigned short* tile,
                                      int m, int sel, int kbase, int ldk) {
#pragma unroll
  for (int i = 0; i < 8; ++i) {
    const int kl = 2 * i + 16 * sel; // lanes 0-15: K0-15; 16-31: K16-31
    a.u[i] = *(const unsigned int*)(tile + m * ldk + kbase + kl);
  }
}

// ---------------------------------------------------------------------------
// K3: GEMM1 + ReLU:  mid = relu(h0b @ W1 + b1)  -> bf16 [32768, 512]
// one block per 16-row M tile; 8 waves x 4 reps cover the 32 N tiles.
// ---------------------------------------------------------------------------
__global__ void __launch_bounds__(256)
k_gemm1(const unsigned short* __restrict__ h0b, const unsigned int* __restrict__ W1p,
        const float* __restrict__ b1, unsigned short* __restrict__ mid) {
  __shared__ unsigned short atile[16 * 256];  // 8 KB
  const int mt = blockIdx.x, tid = threadIdx.x;
  {
    const unsigned int* src = (const unsigned int*)(h0b + (size_t)mt * 16 * 256);
    unsigned int* dst = (unsigned int*)atile;
    for (int i = tid; i < 2048; i += 256) dst[i] = src[i];
  }
  __syncthreads();
  const int wave = tid >> 5, lane = tid & 31;
  const int m = lane & 15, sel = lane >> 4;
#pragma unroll 1
  for (int rep = 0; rep < 4; ++rep) {
    const int nt = wave + rep * 8;
    const int n  = nt * 16 + m;
    const float bias = b1[n];
    v8f acc;
#pragma unroll
    for (int i = 0; i < 8; ++i) acc[i] = bias;
#pragma unroll
    for (int ks = 0; ks < 8; ++ks) {
      BfFrag a, bf;
      loadA(a, atile, m, sel, ks * 32, 256);
      const unsigned int* bp = W1p + ((size_t)(nt * 8 + ks) * 32 + lane) * 8;
#pragma unroll
      for (int i = 0; i < 8; ++i) bf.u[i] = bp[i];
      acc = __builtin_amdgcn_wmma_f32_16x16x32_bf16(false, a.v, false, bf.v,
                                                    (short)0, acc, false, false);
    }
#pragma unroll
    for (int i = 0; i < 8; ++i) {
      const float v = acc[i] > 0.f ? acc[i] : 0.f;
      const int row = mt * 16 + i + 8 * sel;
      mid[(size_t)row * 512 + n] = (unsigned short)f2bf(v);
    }
  }
}

// ---------------------------------------------------------------------------
// K4: GEMM2 + residual + LayerNorm:  h = LN(h0 + mid @ W2 + b2)  (in place)
// ---------------------------------------------------------------------------
__global__ void __launch_bounds__(256)
k_gemm2ln(const unsigned short* __restrict__ mid, const unsigned int* __restrict__ W2p,
          const float* __restrict__ b2, const float* __restrict__ gamma,
          const float* __restrict__ beta, float* __restrict__ h /* h0 in, h out */) {
  __shared__ unsigned short atile[16 * 512];  // 16 KB
  __shared__ float hbuf[16 * 256];            // 16 KB
  const int mt = blockIdx.x, tid = threadIdx.x;
  {
    const unsigned int* src = (const unsigned int*)(mid + (size_t)mt * 16 * 512);
    unsigned int* dst = (unsigned int*)atile;
    for (int i = tid; i < 4096; i += 256) dst[i] = src[i];
  }
  __syncthreads();
  const int wave = tid >> 5, lane = tid & 31;
  const int m = lane & 15, sel = lane >> 4;
#pragma unroll 1
  for (int rep = 0; rep < 2; ++rep) {
    const int nt = wave + rep * 8;
    const int n  = nt * 16 + m;
    const float bias = b2[n];
    v8f acc;
#pragma unroll
    for (int i = 0; i < 8; ++i) acc[i] = bias;
#pragma unroll
    for (int ks = 0; ks < 16; ++ks) {
      BfFrag a, bf;
      loadA(a, atile, m, sel, ks * 32, 512);
      const unsigned int* bp = W2p + ((size_t)(nt * 16 + ks) * 32 + lane) * 8;
#pragma unroll
      for (int i = 0; i < 8; ++i) bf.u[i] = bp[i];
      acc = __builtin_amdgcn_wmma_f32_16x16x32_bf16(false, a.v, false, bf.v,
                                                    (short)0, acc, false, false);
    }
#pragma unroll
    for (int i = 0; i < 8; ++i) hbuf[(i + 8 * sel) * 256 + n] = acc[i];
  }
  __syncthreads();
  // LayerNorm: wave w handles rows 2w, 2w+1 (32 lanes x 8 cols each)
#pragma unroll 1
  for (int rr = 0; rr < 2; ++rr) {
    const int row  = wave * 2 + rr;
    const int grow = mt * 16 + row;
    float x[8];
    float s = 0.f;
#pragma unroll
    for (int j = 0; j < 8; ++j) {
      const int col = lane + 32 * j;
      x[j] = h[(size_t)grow * 256 + col] + hbuf[row * 256 + col];
      s += x[j];
    }
    s = waveRedSum(s);
    const float mu = s * (1.f / 256.f);
    float vs = 0.f;
#pragma unroll
    for (int j = 0; j < 8; ++j) { const float d = x[j] - mu; vs += d * d; }
    vs = waveRedSum(vs);
    const float inv = rsqrtf(vs * (1.f / 256.f) + 1e-5f);
#pragma unroll
    for (int j = 0; j < 8; ++j) {
      const int col = lane + 32 * j;
      h[(size_t)grow * 256 + col] = (x[j] - mu) * inv * gamma[col] + beta[col];
    }
  }
}

// ---------------------------------------------------------------------------
// K5: LRU writer scan — ONE wave32 per batch chain, zero barriers.
// Lane owns 8 contiguous H-columns. Cross-lane only via shfl butterflies
// (bit-identical results in every lane => uniform branches). ds[64] cached in
// 2 regs/lane (only changes on row overwrite, == reference fm@Wd+bd).
// Sequential slot fill => first-free == count; distinct slow ages => FIFO ring.
// Tok loads for the next 4 steps are double-buffered one chunk ahead: h is
// precomputed so loads are independent of the serial state chain (hides L2).
// ---------------------------------------------------------------------------
__global__ void __launch_bounds__(32)
k_scan(const float* __restrict__ h,
       const float* __restrict__ Wg, const float* __restrict__ bg,
       const float* __restrict__ Wd, const float* __restrict__ bd,
       float* __restrict__ sm, float* __restrict__ fmg, int* __restrict__ cnts) {
  const int b = blockIdx.x, lane = threadIdx.x;
  float* fmb = fmg + (size_t)b * kFAST * kH;
  float* smb = sm  + (size_t)b * kSLOW * kH;
  const float4 z = make_float4(0.f, 0.f, 0.f, 0.f);
  for (int i = lane * 4; i < kFAST * kH; i += 128) *(float4*)(fmb + i) = z;
  for (int i = lane * 4; i < kSLOW * kH; i += 128) *(float4*)(smb + i) = z;

  float wgv[8], wdv[8];
#pragma unroll
  for (int j = 0; j < 8; ++j) { wgv[j] = Wg[lane * 8 + j]; wdv[j] = Wd[lane * 8 + j]; }
  const float bgv = bg[0], bdv = bd[0];
  float dsLo = bdv, dsHi = bdv;              // ds[lane], ds[lane+32]
  int fcount = 0, swcount = 0;               // uniform across wave

  const float* hb = h + (size_t)b * kT * kH + lane * 8;
  float tok[2][4][8];
#pragma unroll
  for (int u = 0; u < 4; ++u) {              // preload chunk 0
    const float* p = hb + (size_t)u * kH;
    *(float4*)&tok[0][u][0] = *(const float4*)p;
    *(float4*)&tok[0][u][4] = *(const float4*)(p + 4);
  }

  for (int c = 0; c < 256; ++c) {            // 256 chunks of 4 (steps 0..1020 guarded)
    const int cur = c & 1;
    if (c + 1 < 256) {                       // prefetch next chunk (rows <= 1023, in range)
      const int nt0 = (c + 1) * 4;
#pragma unroll
      for (int u = 0; u < 4; ++u) {
        const float* p = hb + (size_t)(nt0 + u) * kH;
        *(float4*)&tok[cur ^ 1][u][0] = *(const float4*)p;
        *(float4*)&tok[cur ^ 1][u][4] = *(const float4*)(p + 4);
      }
    }
#pragma unroll
    for (int u = 0; u < 4; ++u) {
      const int t = c * 4 + u;
      if (t < kT - 3) {
        const float* tk = tok[cur][u];
        float p1 = 0.f, p2 = 0.f;
#pragma unroll
        for (int j = 0; j < 8; ++j) { p1 += tk[j] * wgv[j]; p2 += tk[j] * wdv[j]; }
        p1 = waveRedSum(p1);                 // identical in all lanes
        p2 = waveRedSum(p2);
        const float wsig = 1.f / (1.f + __expf(-(p1 + bgv)));
        if (wsig >= 0.4f) {                  // uniform branch
          int fidx;
          if (fcount < kFAST) {
            fidx = fcount; ++fcount;
          } else {
            float v = dsLo; int idx = lane;  // argmin(ds), first-min tie-break
            if (dsHi < v) { v = dsHi; idx = lane + 32; }
#pragma unroll
            for (int off = 16; off > 0; off >>= 1) {
              const float ov = __shfl_xor(v, off, 32);
              const int   oi = __shfl_xor(idx, off, 32);
              if (ov < v || (ov == v && oi < idx)) { v = ov; idx = oi; }
            }
            fidx = idx;
            // demote victim row to slow FIFO slot (pre-overwrite value)
            const float* src = fmb + fidx * kH + lane * 8;
            const float4 a0 = *(const float4*)src;
            const float4 a1 = *(const float4*)(src + 4);
            float* dst = smb + (swcount & (kSLOW - 1)) * kH + lane * 8;
            *(float4*)dst = a0; *(float4*)(dst + 4) = a1;
            ++swcount;
          }
          float* rp = fmb + fidx * kH + lane * 8;
          *(float4*)rp       = *(const float4*)&tk[0];
          *(float4*)(rp + 4) = *(const float4*)&tk[4];
          const float dsNew = p2 + bdv;
          if (fidx == lane)      dsLo = dsNew;
          if (fidx == lane + 32) dsHi = dsNew;
        }
      }
    }
  }
  if (lane == 0) { cnts[2 * b] = fcount; cnts[2 * b + 1] = swcount; }
}

// ---------------------------------------------------------------------------
// K6: attention readout per batch: q = h[:,-1]@Wq+bq, masked softmax over
// fast/slow memories, ctx = f_ctx + need_slow * s_ctx.
// ---------------------------------------------------------------------------
__global__ void __launch_bounds__(256)
k_readout(const float* __restrict__ h, const float* __restrict__ Wq,
          const float* __restrict__ bq, const float* __restrict__ sm,
          const float* __restrict__ fmg, const int* __restrict__ cnts,
          float* __restrict__ ctx, float* __restrict__ needOut) {
  __shared__ float hl[kH];
  __shared__ float qbuf[kH];
  __shared__ float red[8];
  __shared__ float fsc[kFAST];
  __shared__ float ssc[kSLOW];
  const int b = blockIdx.x, tid = threadIdx.x;
  const float* fmb = fmg + (size_t)b * kFAST * kH;
  const float* smb = sm  + (size_t)b * kSLOW * kH;
  const int fcount  = cnts[2 * b];
  const int swcount = cnts[2 * b + 1];

  hl[tid] = h[((size_t)b * kT + (kT - 1)) * kH + tid];
  __syncthreads();
  float qv = bq[tid];
  for (int k = 0; k < kH; ++k) qv += hl[k] * Wq[k * kH + tid];
  qbuf[tid] = qv;
  __syncthreads();
  // fast attention
  if (tid < kFAST) {
    float s = NEGV;
    if (tid < fcount) {
      s = 0.f;
      for (int k = 0; k < kH; ++k) s += fmb[tid * kH + k] * qbuf[k];
    }
    fsc[tid] = s;
  }
  __syncthreads();
  if (tid == 0) {
    float mx = fsc[0];
    for (int i = 1; i < kFAST; ++i) mx = fmaxf(mx, fsc[i]);
    red[0] = mx;
  }
  __syncthreads();
  if (tid < kFAST) fsc[tid] = __expf(fsc[tid] - red[0]);
  __syncthreads();
  if (tid == 0) {
    float sum = 0.f, mxe = 0.f;
    for (int i = 0; i < kFAST; ++i) { sum += fsc[i]; mxe = fmaxf(mxe, fsc[i]); }
    red[1] = 1.f / sum;
    red[2] = mxe / sum;                // max_attn
  }
  __syncthreads();
  const float finv = red[1];
  float fctxv = 0.f;
  for (int i = 0; i < kFAST; ++i) fctxv += fsc[i] * finv * fmb[i * kH + tid];
  // slow attention
  const int sused = swcount < kSLOW ? swcount : kSLOW;
  {
    float s = NEGV;
    if (tid < sused) {
      s = 0.f;
      for (int k = 0; k < kH; ++k) s += smb[tid * kH + k] * qbuf[k];
    }
    ssc[tid] = s;
  }
  __syncthreads();
  if (tid == 0) {
    float mx = ssc[0];
    for (int i = 1; i < kSLOW; ++i) mx = fmaxf(mx, ssc[i]);
    red[3] = mx;
  }
  __syncthreads();
  ssc[tid] = __expf(ssc[tid] - red[3]);
  __syncthreads();
  if (tid == 0) {
    float sum = 0.f;
    for (int i = 0; i < kSLOW; ++i) sum += ssc[i];
    red[4] = 1.f / sum;
  }
  __syncthreads();
  const float sinv = red[4];
  float sctxv = 0.f;
  for (int i = 0; i < kSLOW; ++i) sctxv += ssc[i] * sinv * smb[i * kH + tid];
  const float need = (red[2] < 0.3f) ? 1.f : 0.f;
  ctx[b * kH + tid] = fctxv + need * sctxv;
  if (tid == 0) needOut[b] = need;
}

// ---------------------------------------------------------------------------
// K7: logits = ctx @ Wo + bo  via full-fp32 WMMA (16x16x4).
// One wave per N tile of 16, two M tiles (rows 0-15 and 16-31) per wave.
// Wo is streamed once from HBM (bandwidth-bound) — prefetch one group ahead.
// ---------------------------------------------------------------------------
__global__ void __launch_bounds__(32)
k_logits(const float* __restrict__ ctx, const float* __restrict__ Wo,
         const float* __restrict__ bo, float* __restrict__ out) {
  const int nt   = blockIdx.x;             // 0..1999
  const int lane = threadIdx.x;            // 32 lanes, EXEC full
  const int ncol = nt * 16 + (lane & 15);
  const int sel  = lane >> 4;
  const int m0   = lane & 15;
  const float bias = bo[ncol];
  v8f acc0, acc1;
#pragma unroll
  for (int i = 0; i < 8; ++i) { acc0[i] = bias; acc1[i] = bias; }
#pragma unroll 4
  for (int kk = 0; kk < 64; ++kk) {
    if (kk + 4 < 64) {                     // prefetch next unrolled group of Wo rows
      const int kp = (kk + 4) * 4 + 2 * sel;
      __builtin_prefetch(&Wo[(size_t)kp * kV + ncol], 0, 0);
    }
    v2f a0, a1, bf;
#pragma unroll
    for (int i = 0; i < 2; ++i) {
      const int k = kk * 4 + 2 * sel + i;
      a0[i] = ctx[m0 * kH + k];
      a1[i] = ctx[(m0 + 16) * kH + k];
      bf[i] = Wo[(size_t)k * kV + ncol];
    }
    acc0 = __builtin_amdgcn_wmma_f32_16x16x4_f32(false, a0, false, bf,
                                                 (short)0, acc0, false, false);
    acc1 = __builtin_amdgcn_wmma_f32_16x16x4_f32(false, a1, false, bf,
                                                 (short)0, acc1, false, false);
  }
#pragma unroll
  for (int i = 0; i < 8; ++i) {
    const int r = i + 8 * sel;
    out[(size_t)r * kV + ncol]        = acc0[i];
    out[(size_t)(16 + r) * kV + ncol] = acc1[i];
  }
}

// ---------------------------------------------------------------------------
extern "C" void kernel_launch(void* const* d_in, const int* in_sizes, int n_in,
                              void* d_out, int out_size, void* d_ws, size_t ws_size,
                              hipStream_t stream) {
  (void)in_sizes; (void)n_in; (void)out_size; (void)ws_size;
  const int*   seq   = (const int*)  d_in[0];
  const float* embed = (const float*)d_in[1];
  const float* W1    = (const float*)d_in[2];
  const float* b1    = (const float*)d_in[3];
  const float* W2    = (const float*)d_in[4];
  const float* b2    = (const float*)d_in[5];
  const float* gamma = (const float*)d_in[6];
  const float* beta  = (const float*)d_in[7];
  const float* Wg    = (const float*)d_in[8];
  const float* bg    = (const float*)d_in[9];
  const float* Wd    = (const float*)d_in[10];
  const float* bd    = (const float*)d_in[11];
  const float* Wq    = (const float*)d_in[12];
  const float* bq    = (const float*)d_in[13];
  const float* Wo    = (const float*)d_in[14];
  const float* bo    = (const float*)d_in[15];

  char* ws = (char*)d_ws;
  float*          h0f = (float*)(ws + 0);                    // 33554432 B (h0, then h in place)
  unsigned short* h0b = (unsigned short*)(ws + 33554432);    // 16777216 B
  unsigned short* mid = (unsigned short*)(ws + 50331648);    // 33554432 B
  unsigned int*   W1p = (unsigned int*)(ws + 83886080);      //   262144 B
  unsigned int*   W2p = (unsigned int*)(ws + 84148224);      //   262144 B
  float*          smw = (float*)(ws + 84410368);             //  8388608 B
  float*          fmg = (float*)(ws + 92798976);             //  2097152 B
  float*          ctx = (float*)(ws + 94896128);             //    32768 B
  int*            cnt = (int*)(ws + 94928896);               //      256 B

  float* logits = (float*)d_out;
  float* needs  = logits + (size_t)kB * kV;

  hipLaunchKernelGGL(k_gather,  dim3(kB * kT), dim3(kH), 0, stream, seq, embed, h0f, h0b);
  hipLaunchKernelGGL(k_pack,    dim3(512), dim3(256), 0, stream, W1, W2, W1p, W2p);
  hipLaunchKernelGGL(k_gemm1,   dim3(2048), dim3(256), 0, stream, h0b, W1p, b1, mid);
  hipLaunchKernelGGL(k_gemm2ln, dim3(2048), dim3(256), 0, stream, mid, W2p, b2, gamma, beta, h0f);
  hipLaunchKernelGGL(k_scan,    dim3(kB), dim3(32), 0, stream,
                     h0f, Wg, bg, Wd, bd, smw, fmg, cnt);
  hipLaunchKernelGGL(k_readout, dim3(kB), dim3(256), 0, stream,
                     h0f, Wq, bq, smw, fmg, cnt, ctx, needs);
  hipLaunchKernelGGL(k_logits,  dim3(kV / 16), dim3(32), 0, stream, ctx, Wo, bo, logits);
}
